// Autograd4bitQuantLinear_11458972745921
// MI455X (gfx1250) — matrix operand order repaired and checked by reading, not compiled, not run on hardware
//
#include <hip/hip_runtime.h>

// GPTQ 4-bit dequant + GEMM for MI455X (gfx1250, wave32, WMMA)
// out[M=8192, N=4096] = x[M, K=4096] * W[K, N] + bias
// W[k][n] = scales[k/128][n] * (nib(qweight[k/8][n], k%8) - (nib(qzeros[k/128][n/8], n%8) + 1))

typedef __attribute__((ext_vector_type(16))) __bf16 v16bf;
typedef __attribute__((ext_vector_type(8)))  float  v8f;

#define IN_F   4096
#define OUT_F  4096
#define M_TOT  8192
#define BM     128
#define BN     256
#define BK     64
#define LDST   72   // BK + 8 bf16 elements of padding (144B row pitch)

union FragU { v16bf v; uint4 q[2]; };

__device__ __forceinline__ unsigned int pack_bf16(float a, float b) {
    unsigned int ua = __builtin_bit_cast(unsigned int, a);
    unsigned int ub = __builtin_bit_cast(unsigned int, b);
    ua = (ua + 0x7FFFu + ((ua >> 16) & 1u)) >> 16;           // RNE to bf16
    ub = (ub + 0x7FFFu + ((ub >> 16) & 1u)) & 0xFFFF0000u;
    return (ua & 0xFFFFu) | ub;
}

__global__ __launch_bounds__(256)
void gptq_gemm_bf16_wmma(const float* __restrict__ x,
                         const int*   __restrict__ qweight,
                         const int*   __restrict__ qzeros,
                         const float* __restrict__ scales,
                         const float* __restrict__ bias,
                         float*       __restrict__ out)
{
    __shared__ unsigned short As[BM * LDST];   // x tile, bf16, [m][k]          (18 KB)
    __shared__ unsigned short Bs[BN * LDST];   // dequant W tile, bf16, [n][k]  (36 KB)

    const int tid  = threadIdx.x;
    const int lane = tid & 31;
    const int wave = tid >> 5;        // 0..7
    const int wm   = wave >> 2;       // 0..1 : 64-row slab of M
    const int wn   = wave & 3;        // 0..3 : 64-col slab of N
    const int hl   = lane >> 4;       // half-wave select
    const int lm   = lane & 15;

    const int block_n = blockIdx.x * BN;
    const int block_m = blockIdx.y * BM;

    // A staging: thread -> (row, 32-wide k segment)
    const int a_m = tid >> 1;                  // 0..127
    const int a_k = (tid & 1) * 32;            // 0 or 32
    const float* a_ptr = x + (size_t)(block_m + a_m) * IN_F + a_k;

    // B staging: thread -> (packed k-row, 8 consecutive columns = one qzeros word)
    const int b_krow = tid >> 5;               // 0..7  (covers k = krow*8 .. +8)
    const int b_n8   = (tid & 31) * 8;         // 0,8,...,248
    const int n_g    = block_n + b_n8;

    v8f acc[4][4] = {};   // 128 f32 accumulators per lane (64x64 wave tile)

    for (int k0 = 0; k0 < IN_F; k0 += BK) {
        // ---- issue global loads for this chunk (overlap prior chunk's WMMAs) ----
        float4 av[8];
        #pragma unroll
        for (int v = 0; v < 8; ++v)
            av[v] = ((const float4*)(a_ptr + k0))[v];

        const int g = k0 >> 7;  // group (GROUPSIZE=128; constant over a BK=64 chunk)
        const size_t qrow = (size_t)((k0 >> 3) + b_krow) * OUT_F + n_g;
        const int4   qw0 = *(const int4*)(qweight + qrow);
        const int4   qw1 = *(const int4*)(qweight + qrow + 4);
        const float4 sc0 = *(const float4*)(scales + (size_t)g * OUT_F + n_g);
        const float4 sc1 = *(const float4*)(scales + (size_t)g * OUT_F + n_g + 4);
        const unsigned int zword =
            (unsigned int)qzeros[(size_t)g * (OUT_F / 8) + (n_g >> 3)];

        // prefetch next chunk (emits global_prefetch_b8)
        if (k0 + BK < IN_F) {
            __builtin_prefetch(a_ptr + k0 + BK, 0, 1);
            __builtin_prefetch(qweight + (size_t)(((k0 + BK) >> 3) + b_krow) * OUT_F + n_g, 0, 1);
        }

        __syncthreads();  // previous chunk's compute done before LDS overwrite

        // ---- x fp32 -> bf16 into As ----
        {
            uint4 p[4];
            #pragma unroll
            for (int v = 0; v < 4; ++v) {
                p[v].x = pack_bf16(av[2*v].x,   av[2*v].y);
                p[v].y = pack_bf16(av[2*v].z,   av[2*v].w);
                p[v].z = pack_bf16(av[2*v+1].x, av[2*v+1].y);
                p[v].w = pack_bf16(av[2*v+1].z, av[2*v+1].w);
            }
            uint4* dst = (uint4*)&As[a_m * LDST + a_k];
            dst[0] = p[0]; dst[1] = p[1]; dst[2] = p[2]; dst[3] = p[3];
        }

        // ---- dequant 4-bit -> bf16 into Bs ([n][k]) ----
        {
            const unsigned int wq[8] = { (unsigned)qw0.x, (unsigned)qw0.y, (unsigned)qw0.z, (unsigned)qw0.w,
                                         (unsigned)qw1.x, (unsigned)qw1.y, (unsigned)qw1.z, (unsigned)qw1.w };
            const float ss[8] = { sc0.x, sc0.y, sc0.z, sc0.w, sc1.x, sc1.y, sc1.z, sc1.w };
            #pragma unroll
            for (int c = 0; c < 8; ++c) {
                const unsigned int w = wq[c];
                const float s  = ss[c];
                const float o  = -s * (float)(((zword >> (c * 4)) & 0xFu) + 1u);  // -s*(z+1)
                float d[8];
                #pragma unroll
                for (int nib = 0; nib < 8; ++nib)
                    d[nib] = __builtin_fmaf((float)((w >> (nib * 4)) & 0xFu), s, o);
                uint4 packed;
                packed.x = pack_bf16(d[0], d[1]);
                packed.y = pack_bf16(d[2], d[3]);
                packed.z = pack_bf16(d[4], d[5]);
                packed.w = pack_bf16(d[6], d[7]);
                *(uint4*)&Bs[(b_n8 + c) * LDST + b_krow * 8] = packed;
            }
        }

        __syncthreads();

        // ---- two WMMA K-steps over the 64-wide chunk ----
        #pragma unroll
        for (int kk = 0; kk < BK; kk += 32) {
            // B fragments: lanes 0-15 hold K=0..7/16..23, lanes 16-31: +8, column n per lane
            FragU bfr[4];
            #pragma unroll
            for (int j = 0; j < 4; ++j) {
                const int n = wn * 64 + j * 16 + lm;
                const unsigned short* base = &Bs[n * LDST + kk + hl * 8];
                bfr[j].q[0] = *(const uint4*)(base);
                bfr[j].q[1] = *(const uint4*)(base + 16);
            }
            // A fragments: 16-bit A 16x32 layout (lanes 0-15: K=0..7/16..23; lanes 16-31: +8)
            #pragma unroll
            for (int i = 0; i < 4; ++i) {
                FragU afr;
                const int m = wm * 64 + i * 16 + lm;
                const unsigned short* base = &As[m * LDST + kk + hl * 8];
                afr.q[0] = *(const uint4*)(base);
                afr.q[1] = *(const uint4*)(base + 16);
                #pragma unroll
                for (int j = 0; j < 4; ++j)
                    acc[i][j] = __builtin_amdgcn_wmma_f32_16x16x32_bf16(
                        /*neg_a=*/false, afr.v, /*neg_b=*/false, bfr[j].v,
                        /*c_mod=*/(short)0, acc[i][j],
                        /*reuse_a=*/false, /*reuse_b=*/false);
            }
        }
    }

    // ---- epilogue: C layout VGPR r -> rows r (lanes 0-15) and r+8 (lanes 16-31) ----
    #pragma unroll
    for (int j = 0; j < 4; ++j) {
        const int n  = block_n + wn * 64 + j * 16 + lm;
        const float bv = bias[n];
        #pragma unroll
        for (int i = 0; i < 4; ++i) {
            const int mbase = block_m + wm * 64 + i * 16 + hl * 8;
            #pragma unroll
            for (int r = 0; r < 8; ++r)
                out[(size_t)(mbase + r) * OUT_F + n] = acc[i][j][r] + bv;
        }
    }
}

extern "C" void kernel_launch(void* const* d_in, const int* in_sizes, int n_in,
                              void* d_out, int out_size, void* d_ws, size_t ws_size,
                              hipStream_t stream) {
    (void)in_sizes; (void)n_in; (void)out_size; (void)d_ws; (void)ws_size;
    const float* x       = (const float*)d_in[0];
    const int*   qweight = (const int*)d_in[1];
    const int*   qzeros  = (const int*)d_in[2];
    const float* scales  = (const float*)d_in[3];
    const float* bias    = (const float*)d_in[4];
    float*       out     = (float*)d_out;

    dim3 grid(OUT_F / BN, M_TOT / BM, 1);   // 16 x 64 blocks
    dim3 block(256, 1, 1);                  // 8 waves (wave32)
    gptq_gemm_bf16_wmma<<<grid, block, 0, stream>>>(x, qweight, qzeros, scales, bias, out);
}
